// AttentionFusion_85950885528473
// MI455X (gfx1250) — matrix operand order repaired and checked by reading, not compiled, not run on hardware
//
#include <hip/hip_runtime.h>
#include <hip/hip_bf16.h>

#define DEVI __device__ __forceinline__

typedef __bf16 bf16_t;
typedef __attribute__((ext_vector_type(16))) __bf16 bf16x16;
typedef __attribute__((ext_vector_type(8)))  __bf16 bf16x8;
typedef __attribute__((ext_vector_type(8)))  float   f32x8;

constexpr int B_ = 4, C_ = 256, H_ = 64, W_ = 64, N_ = H_ * W_;  // N = 4096

// ---------- helpers ----------

DEVI bf16_t f2bf(float f) {               // round-to-nearest-even fp32 -> bf16
  unsigned u = __builtin_bit_cast(unsigned, f);
  u += 0x7FFFu + ((u >> 16) & 1u);
  unsigned short h = (unsigned short)(u >> 16);
  return __builtin_bit_cast(bf16_t, h);
}

DEVI f32x8 zero8() {
  f32x8 z;
#pragma unroll
  for (int i = 0; i < 8; ++i) z[i] = 0.f;
  return z;
}

DEVI f32x8 wmma_bf16(bf16x16 a, bf16x16 b, f32x8 c) {
  // (neg_a, A, neg_b, B, c_mod, C, reuse_a, reuse_b)
  return __builtin_amdgcn_wmma_f32_16x16x32_bf16(false, a, false, b, (short)0, c,
                                                 false, false);
}

DEVI bf16x16 cat16(bf16x8 lo, bf16x8 hi) {
  bf16x16 r;
#pragma unroll
  for (int i = 0; i < 8; ++i) { r[i] = lo[i]; r[i + 8] = hi[i]; }
  return r;
}

// A fragment (16x32 bf16): storage [M][K] row-major, base pre-offset to tile row 0.
// lane<16: K = k0+{0..7, 16..23}; lane>=16: K = k0+{8..15, 24..31}; row = lane&15.
DEVI bf16x16 load_a_frag(const bf16_t* base, int ld, int k0) {
  int lane = threadIdx.x & 31;
  const bf16_t* p = base + (long)(lane & 15) * ld + k0 + ((lane >> 4) << 3);
  return cat16(*(const bf16x8*)p, *(const bf16x8*)(p + 16));
}

// B fragment (32x16 bf16): storage [Ncol][K] (each output column's K-run contiguous).
// lane<16: K = k0+{0..7, 16..23}; lane>=16: K = k0+{8..15, 24..31}; col = col0 + (lane&15).
DEVI bf16x16 load_b_frag(const bf16_t* base, int ld, int col0, int k0) {
  int lane = threadIdx.x & 31;
  const bf16_t* p = base + (long)(col0 + (lane & 15)) * ld + k0 + ((lane >> 4) << 3);
  return cat16(*(const bf16x8*)p, *(const bf16x8*)(p + 16));
}

// 16-byte global -> LDS async copy (CDNA5 GLOBAL_LOAD_ASYNC_TO_LDS_B128, ASYNCcnt).
// Low 32 bits of a generic LDS pointer are the LDS byte address (ISA 10.2).
DEVI void cp16(bf16_t* ldst, const bf16_t* gsrc) {
  unsigned lds_addr = (unsigned)(unsigned long long)(void*)ldst;
  asm volatile("global_load_async_to_lds_b128 %0, %1, off"
               :
               : "v"(lds_addr), "v"(gsrc)
               : "memory");
}

DEVI void cp_wait() {
  asm volatile("s_wait_asynccnt 0x0" ::: "memory");
}

DEVI float rmax16(float v) {
#pragma unroll
  for (int m = 1; m < 16; m <<= 1) v = fmaxf(v, __shfl_xor(v, m, 32));
  return v;
}
DEVI float rsum16(float v) {
#pragma unroll
  for (int m = 1; m < 16; m <<= 1) v += __shfl_xor(v, m, 32);
  return v;
}

// ---------- kernel 1: fp32 [B][C][N] -> bf16 [B][N][C] transpose ----------

__global__ void k_cvt_x(const float* __restrict__ x, bf16_t* __restrict__ xt) {
  long i = (long)blockIdx.x * blockDim.x + threadIdx.x;   // over B*C*N
  int n = (int)(i & (N_ - 1));
  int c = (int)((i >> 12) & (C_ - 1));
  int b = (int)(i >> 20);
  xt[((long)b * N_ + n) * C_ + c] = f2bf(x[i]);
}

__global__ void k_cvt_w(const float* __restrict__ w, bf16_t* __restrict__ o, int n) {
  int i = blockIdx.x * blockDim.x + threadIdx.x;
  if (i < n) o[i] = f2bf(w[i]);
}

// ---------- kernel 2: QKV projection ----------
// q[b][n][co], k[b][n][co] (position-major), v[b][co][n] (channel-major)

__global__ void __launch_bounds__(256, 1)
k_qkv(const bf16_t* __restrict__ xqT, const bf16_t* __restrict__ xkT,
      const bf16_t* __restrict__ Wq, const float* __restrict__ bq,
      const bf16_t* __restrict__ Wk, const float* __restrict__ bk,
      const bf16_t* __restrict__ Wv, const float* __restrict__ bv,
      bf16_t* __restrict__ qo, bf16_t* __restrict__ ko, bf16_t* __restrict__ vo) {
  const int wave = threadIdx.x >> 5, lane = threadIdx.x & 31;
  const int l16 = lane & 15, lh = lane >> 4;
  const int bb = blockIdx.y;
  const int n0 = blockIdx.x * 128 + wave * 16;

  const bf16_t* xq = xqT + ((long)bb * N_ + n0) * C_;
  const bf16_t* xk = xkT + ((long)bb * N_ + n0) * C_;

  bf16x16 aq[8], ak[8];
#pragma unroll
  for (int kc = 0; kc < 8; ++kc) {
    aq[kc] = load_a_frag(xq, C_, kc * 32);
    ak[kc] = load_a_frag(xk, C_, kc * 32);
  }

#pragma unroll 1
  for (int co0 = 0; co0 < C_; co0 += 64) {
    f32x8 accq[4], acck[4], accv[4];
#pragma unroll
    for (int j = 0; j < 4; ++j) { accq[j] = zero8(); acck[j] = zero8(); accv[j] = zero8(); }
#pragma unroll
    for (int kc = 0; kc < 8; ++kc) {
#pragma unroll
      for (int j = 0; j < 4; ++j) {
        accq[j] = wmma_bf16(aq[kc], load_b_frag(Wq, C_, co0 + j * 16, kc * 32), accq[j]);
        acck[j] = wmma_bf16(ak[kc], load_b_frag(Wk, C_, co0 + j * 16, kc * 32), acck[j]);
        accv[j] = wmma_bf16(ak[kc], load_b_frag(Wv, C_, co0 + j * 16, kc * 32), accv[j]);
      }
    }
#pragma unroll
    for (int j = 0; j < 4; ++j) {
      int co = co0 + j * 16 + l16;
      float biq = bq[co], bik = bk[co], biv = bv[co];
#pragma unroll
      for (int r = 0; r < 8; ++r) {
        int n = n0 + lh * 8 + r;                 // D row = (lane/16)*8 + vgpr
        qo[((long)bb * N_ + n) * C_ + co] = f2bf(accq[j][r] + biq);
        ko[((long)bb * N_ + n) * C_ + co] = f2bf(acck[j][r] + bik);
        vo[((long)bb * C_ + co) * N_ + n] = f2bf(accv[j][r] + biv);  // transposed store
      }
    }
  }
}

// ---------- kernel 3: flash attention ----------
// Workgroup = 8 waves x 16 query rows (128 rows). Key blocks of 64 staged once
// per workgroup into LDS via async global->LDS copies, shared by all waves.

constexpr int KLD = 264;  // LDS stride for K block rows [64][KLD]
constexpr int VLD = 72;   // LDS stride for V block rows [256][VLD]
constexpr int PLD = 72;   // per-wave P scratch [16][PLD]
constexpr size_t ATTN_LDS_ELEMS = (size_t)64 * KLD + 256 * VLD + 8 * 16 * PLD;
constexpr size_t ATTN_LDS_BYTES = ATTN_LDS_ELEMS * sizeof(bf16_t);

__global__ void __launch_bounds__(256, 1)
k_attn(const bf16_t* __restrict__ qm, const bf16_t* __restrict__ km,
       const bf16_t* __restrict__ vm, bf16_t* __restrict__ avm) {
  extern __shared__ __align__(16) char smem_raw[];
  bf16_t* kbuf = (bf16_t*)smem_raw;            // [64][KLD]   K block (m-major)
  bf16_t* vbuf = kbuf + 64 * KLD;              // [256][VLD]  V block (c-major)
  bf16_t* pbuf = vbuf + 256 * VLD;             // [8][16][PLD] per-wave P

  const int t = threadIdx.x;
  const int wave = t >> 5, lane = t & 31;
  const int l16 = lane & 15, lh = lane >> 4;
  const int bb = blockIdx.y;
  const int n0 = blockIdx.x * 128 + wave * 16;

  const bf16_t* qbase = qm + ((long)bb * N_ + n0) * C_;
  const bf16_t* kbase = km + (long)bb * N_ * C_;
  const bf16_t* vbase = vm + (long)bb * C_ * N_;
  bf16_t* pw = pbuf + wave * 16 * PLD;

  bf16x16 aq[8];
#pragma unroll
  for (int kc = 0; kc < 8; ++kc) aq[kc] = load_a_frag(qbase, C_, kc * 32);

  f32x8 o[16];                        // O accum: 16 q-rows x 256 channels (fp32)
#pragma unroll
  for (int cf = 0; cf < 16; ++cf) o[cf] = zero8();
  float mrow[8], lrow[8];
#pragma unroll
  for (int r = 0; r < 8; ++r) { mrow[r] = -1e30f; lrow[r] = 0.f; }

#pragma unroll 1
  for (int m0 = 0; m0 < N_; m0 += 64) {
    // ---- stage K block [64][256] and V block [256][64] into LDS (16B chunks) ----
#pragma unroll
    for (int it = 0; it < 8; ++it) {            // K: 2048 chunks / 256 threads
      int ch = it * 256 + t;
      int m  = ch >> 5;                         // 0..63
      int co = (ch & 31) * 8;                   // 0..248
      cp16(kbuf + m * KLD + co, kbase + (long)(m0 + m) * C_ + co);
    }
#pragma unroll
    for (int it = 0; it < 8; ++it) {            // V: 2048 chunks / 256 threads
      int ch = it * 256 + t;
      int c  = ch >> 3;                         // 0..255
      int mo = (ch & 7) * 8;                    // 0..56
      cp16(vbuf + c * VLD + mo, vbase + (long)c * N_ + m0 + mo);
    }
    cp_wait();
    __syncthreads();

    // prefetch next key/value block toward L2 while we compute on this one
    if (m0 + 64 < N_) {
      __builtin_prefetch(kbase + (long)(m0 + 64 + (t >> 2)) * C_ + (t & 3) * 64, 0, 1);
      __builtin_prefetch(vbase + (long)t * N_ + m0 + 64, 0, 1);
    }

    // ---- S = Q . K^T  (16 x 64, accumulate over 256 channels, K from LDS) ----
    f32x8 s[4];
#pragma unroll
    for (int j = 0; j < 4; ++j) s[j] = zero8();
#pragma unroll
    for (int kc = 0; kc < 8; ++kc)
#pragma unroll
      for (int j = 0; j < 4; ++j)
        s[j] = wmma_bf16(aq[kc], load_b_frag(kbuf, KLD, j * 16, kc * 32), s[j]);

    // ---- online softmax (row stats redundant within each 16-lane group) ----
    float alpha[8];
#pragma unroll
    for (int r = 0; r < 8; ++r) {
      float mx = mrow[r];
#pragma unroll
      for (int j = 0; j < 4; ++j) mx = fmaxf(mx, s[j][r]);
      mx = rmax16(mx);
      alpha[r] = __expf(mrow[r] - mx);
      mrow[r] = mx;
    }
#pragma unroll
    for (int j = 0; j < 4; ++j)
#pragma unroll
      for (int r = 0; r < 8; ++r) s[j][r] = __expf(s[j][r] - mrow[r]);
#pragma unroll
    for (int r = 0; r < 8; ++r) {
      float sm = rsum16(s[0][r] + s[1][r] + s[2][r] + s[3][r]);
      lrow[r] = lrow[r] * alpha[r] + sm;
    }
#pragma unroll
    for (int cf = 0; cf < 16; ++cf)
#pragma unroll
      for (int r = 0; r < 8; ++r) o[cf][r] *= alpha[r];

    // ---- P (D layout) -> per-wave LDS -> A layout ----
#pragma unroll
    for (int j = 0; j < 4; ++j)
#pragma unroll
      for (int r = 0; r < 8; ++r)
        pw[(lh * 8 + r) * PLD + j * 16 + l16] = f2bf(s[j][r]);
    __syncthreads();

    // ---- O += P . V  (K-dim = 64 key positions, N-dim = 256 channels) ----
#pragma unroll
    for (int mc = 0; mc < 2; ++mc) {
      bf16x16 ap = load_a_frag(pw, PLD, mc * 32);
#pragma unroll
      for (int cf = 0; cf < 16; ++cf)
        o[cf] = wmma_bf16(ap, load_b_frag(vbuf, VLD, cf * 16, mc * 32), o[cf]);
    }
    __syncthreads();   // all waves done with kbuf/vbuf before next staging
  }

  float inv[8];
#pragma unroll
  for (int r = 0; r < 8; ++r) inv[r] = 1.f / lrow[r];
#pragma unroll
  for (int cf = 0; cf < 16; ++cf)
#pragma unroll
    for (int r = 0; r < 8; ++r) {
      int n = n0 + lh * 8 + r;
      avm[((long)bb * N_ + n) * C_ + cf * 16 + l16] = f2bf(o[cf][r] * inv[r]);
    }
}

// ---------- kernel 4: fused MLP  out = W2.relu(W1.av + b1) + b2 ----------

__global__ void __launch_bounds__(128, 1)
k_mlp(const bf16_t* __restrict__ avm,
      const bf16_t* __restrict__ W1b, const float* __restrict__ b1,
      const bf16_t* __restrict__ W2b, const float* __restrict__ b2,
      float* __restrict__ out) {
  __shared__ __align__(16) bf16_t hbuf[4][16][264];  // per-wave hidden tile (D->A)
  const int wave = threadIdx.x >> 5, lane = threadIdx.x & 31;
  const int l16 = lane & 15, lh = lane >> 4;
  const int bb = blockIdx.y;
  const int n0 = blockIdx.x * 64 + wave * 16;

  const bf16_t* abase = avm + ((long)bb * N_ + n0) * C_;
  bf16x16 aa[8];
#pragma unroll
  for (int kc = 0; kc < 8; ++kc) aa[kc] = load_a_frag(abase, C_, kc * 32);

#pragma unroll 1
  for (int h0 = 0; h0 < C_; h0 += 64) {
    f32x8 acc[4];
#pragma unroll
    for (int j = 0; j < 4; ++j) acc[j] = zero8();
#pragma unroll
    for (int kc = 0; kc < 8; ++kc)
#pragma unroll
      for (int j = 0; j < 4; ++j)
        acc[j] = wmma_bf16(aa[kc], load_b_frag(W1b, C_, h0 + j * 16, kc * 32), acc[j]);
#pragma unroll
    for (int j = 0; j < 4; ++j) {
      float bi = b1[h0 + j * 16 + l16];
#pragma unroll
      for (int r = 0; r < 8; ++r)
        hbuf[wave][lh * 8 + r][h0 + j * 16 + l16] = f2bf(fmaxf(acc[j][r] + bi, 0.f));
    }
  }
  __syncthreads();

#pragma unroll 1
  for (int co0 = 0; co0 < C_; co0 += 64) {
    f32x8 acc[4];
#pragma unroll
    for (int j = 0; j < 4; ++j) acc[j] = zero8();
#pragma unroll
    for (int kc = 0; kc < 8; ++kc) {
      bf16x16 ah = load_a_frag(&hbuf[wave][0][0], 264, kc * 32);
#pragma unroll
      for (int j = 0; j < 4; ++j)
        acc[j] = wmma_bf16(ah, load_b_frag(W2b, C_, co0 + j * 16, kc * 32), acc[j]);
    }
#pragma unroll
    for (int j = 0; j < 4; ++j) {
      int co = co0 + j * 16 + l16;
      float bi = b2[co];
#pragma unroll
      for (int r = 0; r < 8; ++r) {
        int n = n0 + lh * 8 + r;
        out[((long)bb * C_ + co) * N_ + n] = acc[j][r] + bi;  // [B][Co][H][W] fp32
      }
    }
  }
}

// ---------- launch ----------

extern "C" void kernel_launch(void* const* d_in, const int* in_sizes, int n_in,
                              void* d_out, int out_size, void* d_ws, size_t ws_size,
                              hipStream_t stream) {
  (void)in_sizes; (void)n_in; (void)out_size; (void)ws_size;
  const size_t NC = (size_t)B_ * N_ * C_;  // 4,194,304 elements (8 MB as bf16)

  bf16_t* xqT = (bf16_t*)d_ws;
  bf16_t* xkT = xqT + NC;
  bf16_t* qm  = xkT + NC;
  bf16_t* km  = qm + NC;
  bf16_t* vm  = km + NC;
  bf16_t* avm = vm + NC;
  bf16_t* Wqb = avm + NC;
  bf16_t* Wkb = Wqb + C_ * C_;
  bf16_t* Wvb = Wkb + C_ * C_;
  bf16_t* W1b = Wvb + C_ * C_;
  bf16_t* W2b = W1b + C_ * C_;

  k_cvt_x<<<(int)(NC / 256), 256, 0, stream>>>((const float*)d_in[0], xqT);
  k_cvt_x<<<(int)(NC / 256), 256, 0, stream>>>((const float*)d_in[1], xkT);
  k_cvt_w<<<C_ * C_ / 256, 256, 0, stream>>>((const float*)d_in[2],  Wqb, C_ * C_);
  k_cvt_w<<<C_ * C_ / 256, 256, 0, stream>>>((const float*)d_in[4],  Wkb, C_ * C_);
  k_cvt_w<<<C_ * C_ / 256, 256, 0, stream>>>((const float*)d_in[6],  Wvb, C_ * C_);
  k_cvt_w<<<C_ * C_ / 256, 256, 0, stream>>>((const float*)d_in[8],  W1b, C_ * C_);
  k_cvt_w<<<C_ * C_ / 256, 256, 0, stream>>>((const float*)d_in[10], W2b, C_ * C_);

  k_qkv<<<dim3(N_ / 128, B_), 256, 0, stream>>>(
      xqT, xkT, Wqb, (const float*)d_in[3], Wkb, (const float*)d_in[5],
      Wvb, (const float*)d_in[7], qm, km, vm);

  k_attn<<<dim3(N_ / 128, B_), 256, (int)ATTN_LDS_BYTES, stream>>>(qm, km, vm, avm);

  k_mlp<<<dim3(N_ / 64, B_), 128, 0, stream>>>(
      avm, W1b, (const float*)d_in[9], W2b, (const float*)d_in[11], (float*)d_out);
}
